// TokenEmbedding_78340203479166
// MI455X (gfx1250) — matrix operand order repaired and checked by reading, not compiled, not run on hardware
//
#include <hip/hip_runtime.h>

// ---------------------------------------------------------------------------
// Problem constants (from reference):
//  x:[512,3,3,128,64] f32, W0:[192,1024], b0:[1024], W1:[1024,256], b1:[256]
//  out:[512,3,128,256] f32
// ---------------------------------------------------------------------------
#define BATCH 512
#define NBLK  3
#define NCH   3
#define NTOK  128
#define NBIN  64
#define IN_SZ 192
#define HID   1024
#define EMB   256

typedef __attribute__((ext_vector_type(16))) __bf16        v16bf;
typedef __attribute__((ext_vector_type(8)))  __bf16        v8bf;
typedef __attribute__((ext_vector_type(8)))  float         v8f;
typedef __attribute__((ext_vector_type(8)))  float         v8fv;
typedef __attribute__((ext_vector_type(4)))  unsigned int  v4u;
typedef __attribute__((ext_vector_type(8)))  int           v8i;
typedef __attribute__((ext_vector_type(4)))  int           v4i;

#define PITCH_H 136                    // 128 + 8 bf16 pad -> 272B rows, 16B aligned
#define W0P_BYTES (IN_SZ * HID * 2)    // 384 KB packed bf16 W0 in d_ws

// LDS partition (dynamic shared memory, offsets double as TDM lds_addr)
#define HBUF_OFF   0
#define HBUF_BYTES (NTOK * PITCH_H * 2)        // 34816
#define W0S_OFF    HBUF_BYTES                  // 34816
#define W0S_BYTES  (6 * 8 * 1024)              // 49152  (6 K-tiles x 8 N-tiles)
#define W1S_OFF    (W0S_OFF + W0S_BYTES)       // 83968
#define W1S_BYTES  (4 * 16 * 1024)             // 65536  (4 K-tiles x 16 N-tiles)
#define SMEM_BYTES (W1S_OFF + W1S_BYTES)       // 149504 (<160KB -> 2 WGs/WGP)

// Toolchain split: amdgpu-toolchain ships the TDM header and uses the 6-arg
// builtin; ROCm 7.2 uses the 5-arg form.
#if __has_include(<hip/amd_detail/amd_gfx1250_TDM.h>)
#define TDM_ARITY6 1
#endif

static __device__ inline v8f vzero() {
    v8f z;
#pragma unroll
    for (int i = 0; i < 8; ++i) z[i] = 0.0f;
    return z;
}

static __device__ inline v16bf cat8(v8bf lo, v8bf hi) {
    return __builtin_shufflevector(lo, hi, 0, 1, 2, 3, 4, 5, 6, 7,
                                           8, 9, 10, 11, 12, 13, 14, 15);
}

// A-fragment (16x32 bf16) from row-major fp32: lane half picks K offsets
// {0..7,16..23} / {8..15,24..31}; p0/p1 are the two 8-float chunk bases.
static __device__ inline v16bf load_a_f32(const float* p0, const float* p1) {
    v8fv f0 = *(const v8fv*)p0;   // 2x global_load_b128
    v8fv f1 = *(const v8fv*)p1;
    v16bf a;
#pragma unroll
    for (int i = 0; i < 8; ++i) {
        a[i]     = (__bf16)f0[i];
        a[i + 8] = (__bf16)f1[i];
    }
    return a;
}

static __device__ inline v8f wmma_bf16(v16bf a, v16bf b, v8f c) {
    return __builtin_amdgcn_wmma_f32_16x16x32_bf16(
        /*neg_a=*/false, a, /*neg_b=*/false, b,
        /*c_mod=*/(short)0, c, /*reuse_a=*/false, /*reuse_b=*/false);
}

// ---------------------------------------------------------------------------
// Tensor Data Mover: issue a 2D tile load (global -> LDS).
// Descriptor packing per CDNA5 ISA ch.8 (D# groups 0/1; groups 2/3 zero).
// Units: data_size = 8 bytes (code 3); td*/tl*/stride0 are in 8-byte units.
// ---------------------------------------------------------------------------
static __device__ inline void tdm_load_2d(unsigned long long ga, unsigned lds_off,
                                          unsigned td0, unsigned td1,
                                          unsigned tl0, unsigned tl1,
                                          unsigned long long stride0) {
    v4u g0;
    g0[0] = 1u;                                     // count=1, user mode, no gather
    g0[1] = lds_off;                                // lds_addr (bytes)
    g0[2] = (unsigned)(ga & 0xFFFFFFFFu);           // global_addr[31:0]
    g0[3] = (unsigned)((ga >> 32) & 0x1FFFFFFu)     // global_addr[56:32]
          | (2u << 30);                             // type = 2 ("image")
    v8i g1;
    g1[0] = (int)(3u << 16);                        // wg_mask=0, data_size=8B, flags=0
    g1[1] = (int)((td0 & 0xFFFFu) << 16);           // tensor_dim0[15:0] (atomic addr=0)
    g1[2] = (int)((td0 >> 16) | ((td1 & 0xFFFFu) << 16));   // td0[31:16] | td1[15:0]
    g1[3] = (int)((td1 >> 16) | (tl0 << 16));       // td1[31:16] | tile_dim0
    g1[4] = (int)(tl1 & 0xFFFFu);                   // tile_dim1 (tile_dim2 = 0)
    g1[5] = (int)(unsigned)(stride0 & 0xFFFFFFFFu); // tensor_dim0_stride[31:0]
    g1[6] = (int)(unsigned)(stride0 >> 32);         // stride0[47:32] (stride1 = 0)
    g1[7] = 0;
    v4i z4 = {0, 0, 0, 0};
#ifdef TDM_ARITY6
    v8i z8 = {0, 0, 0, 0, 0, 0, 0, 0};
    __builtin_amdgcn_tensor_load_to_lds(g0, g1, z4, z4, z8, 0);
#else
    __builtin_amdgcn_tensor_load_to_lds(g0, g1, z4, z4, 0);
#endif
}

// ---------------------------------------------------------------------------
// Weight packer: fp32 row-major W[K x N] -> bf16 WMMA-B fragment tiles.
// Tile (kt, nt) = K rows [kt*32,+32), N cols [nt*16,+16); 1024B per tile,
// lane l's 16 bf16 contiguous at l*32B (half = l>>4 -> K {0..7,16..23}/{8..15,24..31}).
// ---------------------------------------------------------------------------
__global__ __launch_bounds__(256) void pack_w_kernel(const float* __restrict__ W,
                                                     __bf16* __restrict__ ws,
                                                     int ncols, int ntiles_n,
                                                     int total_tiles) {
    int gid  = blockIdx.x * blockDim.x + threadIdx.x;
    int tile = gid >> 5;
    if (tile >= total_tiles) return;
    int lane = gid & 31;
    int kt   = tile / ntiles_n;
    int nt   = tile % ntiles_n;
    int half = lane >> 4;
    int n    = nt * 16 + (lane & 15);
    __bf16* p = ws + (size_t)tile * 512 + (size_t)lane * 16;
#pragma unroll
    for (int v = 0; v < 8; ++v) {
        int k = kt * 32 + half * 8 + 2 * v + (v >= 4 ? 8 : 0);
        p[2 * v]     = (__bf16)W[(size_t)k * ncols + n];
        p[2 * v + 1] = (__bf16)W[(size_t)(k + 1) * ncols + n];
    }
}

// ---------------------------------------------------------------------------
// Fused 2-layer MLP. One workgroup per (batch, block) pair = 128 token rows.
// Per L0 N-chunk (=L1 K-chunk) of 128: wave 0 TDM-stages the W0 chunk tiles
// (2D: 6 rows x 8KB, 64KB row stride) and the W1 K-slab (64KB linear) into
// LDS, waits TENSORcnt, then all 8 waves run WMMA with B-fragments from LDS.
// ---------------------------------------------------------------------------
__global__ __launch_bounds__(256) void mlp_fused_kernel(
    const float* __restrict__ x, const __bf16* __restrict__ w0p,
    const float* __restrict__ b0, const __bf16* __restrict__ w1p,
    const float* __restrict__ b1, float* __restrict__ out) {

    const int tid    = threadIdx.x;
    const int lane   = tid & 31;
    const int wave   = tid >> 5;
    const int mgroup = wave >> 1;           // 0..3
    const int nhalf  = wave & 1;            // 0..1
    const int mbase  = mgroup * 32;

    const int rowq = lane & 15;             // A-frag: lanes carry M=0..15
    const int koff = (lane >> 4) * 8;       // A-frag K offset per lane half
    const int colq = lane & 15;             // C-tile: lanes carry N=0..15
    const int rsel = (lane >> 4) * 8;       // C-tile: lane half -> rows 0-7/8-15

    const long pair = blockIdx.x;                         // b*NBLK + blk
    const float* xb = x + pair * (long)(NCH * NTOK * NBIN);
    float* ob       = out + pair * (long)(NTOK * EMB);

    extern __shared__ char smem[];
    __bf16*     hbuf = (__bf16*)(smem + HBUF_OFF);
    const char* w0s  = smem + W0S_OFF;
    const char* w1s  = smem + W1S_OFF;

    v8f acc1[2][8];
#pragma unroll
    for (int ms = 0; ms < 2; ++ms)
#pragma unroll
        for (int n = 0; n < 8; ++n) acc1[ms][n] = vzero();

    for (int c = 0; c < 8; ++c) {           // L0 N-chunk == L1 K-chunk (128)
        // ---- TDM stage: W0 chunk (6x8KB, stride 64KB) + W1 slab (64KB) ----
        if (wave == 0) {
            tdm_load_2d((unsigned long long)w0p + (unsigned long long)c * 8192ull,
                        W0S_OFF, /*td0=*/8192, /*td1=*/6,
                        /*tl0=*/1024, /*tl1=*/6, /*stride0=*/8192ull);
            tdm_load_2d((unsigned long long)w1p + (unsigned long long)c * 65536ull,
                        W1S_OFF, /*td0=*/8192, /*td1=*/1,
                        /*tl0=*/8192, /*tl1=*/1, /*stride0=*/8192ull);
            __builtin_amdgcn_s_wait_tensorcnt(0);
        }
        __syncthreads();                    // staged weights visible to all waves

        // ---------------- Layer 0: h_chunk = relu(x @ W0[:, chunk] + b0) ----
        v8f acc0[2][4];
#pragma unroll
        for (int ms = 0; ms < 2; ++ms)
#pragma unroll
            for (int n = 0; n < 4; ++n) acc0[ms][n] = vzero();

#pragma unroll
        for (int kt = 0; kt < 6; ++kt) {    // K = 192 = 6 x 32
            v16bf a[2];
#pragma unroll
            for (int ms = 0; ms < 2; ++ms) {
                int row = mbase + ms * 16 + rowq;         // token index
                int k0  = kt * 32 + koff;                 // feature = ch*64+bin
                int k1  = k0 + 16;
                // x layout [ch][tok][bin]; each 8-elem K chunk stays in one ch
                const float* p0 = xb + (k0 >> 6) * (NTOK * NBIN) + row * NBIN + (k0 & 63);
                const float* p1 = xb + (k1 >> 6) * (NTOK * NBIN) + row * NBIN + (k1 & 63);
                a[ms] = load_a_f32(p0, p1);
            }
#pragma unroll
            for (int n = 0; n < 4; ++n) {
                int ntl = nhalf * 4 + n;                  // chunk-local N-tile 0..7
                const v8bf* bp = (const v8bf*)(w0s +
                                 (size_t)(kt * 8 + ntl) * 1024 + (size_t)lane * 32);
                v16bf bf = cat8(bp[0], bp[1]);            // 2x ds_load_b128
                acc0[0][n] = wmma_bf16(a[0], bf, acc0[0][n]);
                acc0[1][n] = wmma_bf16(a[1], bf, acc0[1][n]);
            }
        }

        // bias + relu -> LDS h (bf16), chunk-local column index
#pragma unroll
        for (int ms = 0; ms < 2; ++ms) {
#pragma unroll
            for (int n = 0; n < 4; ++n) {
                int ncg  = c * 128 + nhalf * 64 + n * 16 + colq; // global col
                int ncc  = nhalf * 64 + n * 16 + colq;           // chunk col
                float bias = b0[ncg];
                int rb = mbase + ms * 16 + rsel;
#pragma unroll
                for (int v = 0; v < 8; ++v) {
                    float val = fmaxf(acc0[ms][n][v] + bias, 0.0f);
                    hbuf[(rb + v) * PITCH_H + ncc] = (__bf16)val;
                }
            }
        }
        __syncthreads();                    // h chunk complete

        // ---------------- Layer 1 partial: acc1 += h_chunk @ W1[chunk,:] ----
#pragma unroll
        for (int kk = 0; kk < 4; ++kk) {    // 128 = 4 x 32
            v16bf a[2];
#pragma unroll
            for (int ms = 0; ms < 2; ++ms) {
                int row = mbase + ms * 16 + rowq;
                const __bf16* hp = &hbuf[row * PITCH_H + kk * 32 + koff];
                a[ms] = cat8(*(const v8bf*)hp, *(const v8bf*)(hp + 16));
            }
#pragma unroll
            for (int n = 0; n < 8; ++n) {
                int ntile = nhalf * 8 + n;
                const v8bf* bp = (const v8bf*)(w1s +
                                 (size_t)(kk * 16 + ntile) * 1024 + (size_t)lane * 32);
                v16bf bf = cat8(bp[0], bp[1]);
                acc1[0][n] = wmma_bf16(a[0], bf, acc1[0][n]);
                acc1[1][n] = wmma_bf16(a[1], bf, acc1[1][n]);
            }
        }
        __syncthreads();                    // all reads of w0s/w1s/hbuf done
    }

    // ---------------- Final epilogue: bias + relu -> out (fp32) -------------
#pragma unroll
    for (int ms = 0; ms < 2; ++ms) {
#pragma unroll
        for (int n = 0; n < 8; ++n) {
            int ncol = nhalf * 128 + n * 16 + colq;
            float bias = b1[ncol];
            int rb = mbase + ms * 16 + rsel;
#pragma unroll
            for (int v = 0; v < 8; ++v) {
                float val = fmaxf(acc1[ms][n][v] + bias, 0.0f);
                ob[(long)(rb + v) * EMB + ncol] = val;
            }
        }
    }
}

// ---------------------------------------------------------------------------
extern "C" void kernel_launch(void* const* d_in, const int* in_sizes, int n_in,
                              void* d_out, int out_size, void* d_ws, size_t ws_size,
                              hipStream_t stream) {
    (void)in_sizes; (void)n_in; (void)out_size; (void)ws_size;
    const float* x  = (const float*)d_in[0];
    const float* W0 = (const float*)d_in[1];
    const float* b0 = (const float*)d_in[2];
    const float* W1 = (const float*)d_in[3];
    const float* b1 = (const float*)d_in[4];
    float* out = (float*)d_out;

    __bf16* w0p = (__bf16*)d_ws;                              // 384 KB
    __bf16* w1p = (__bf16*)((char*)d_ws + W0P_BYTES);         // 512 KB

    // Pack weights fp32 -> bf16 WMMA-B tiles (runs every call; deterministic).
    pack_w_kernel<<<(384 * 32 + 255) / 256, 256, 0, stream>>>(W0, w0p, HID, 64, 384);
    pack_w_kernel<<<(512 * 32 + 255) / 256, 256, 0, stream>>>(W1, w1p, EMB, 16, 512);

    // One workgroup per (batch, block) pair: 512*3 = 1536 workgroups.
    mlp_fused_kernel<<<BATCH * NBLK, 256, SMEM_BYTES, stream>>>(x, w0p, b0, w1p, b1, out);
}
